// PointNetSetAbstraction_11433202942869
// MI455X (gfx1250) — matrix operand order repaired and checked by reading, not compiled, not run on hardware
//
#include <hip/hip_runtime.h>
#include <hip/hip_bf16.h>

typedef __attribute__((ext_vector_type(16))) __bf16 v16bf;
typedef __attribute__((ext_vector_type(8)))  __bf16 v8bf;
typedef __attribute__((ext_vector_type(8)))  float  v8f;

#define B_       16
#define N_       4096
#define NPOINT_  1024
#define NSAMPLE_ 32
#define NTOT     (B_ * NPOINT_ * NSAMPLE_)   /* 524288 samples */
#define NTILES   (NTOT / 16)                 /* 32768 column tiles */
#define K0_      96                          /* 67 padded to multiple of 32 */
#define EPS_     1e-5f
#define R2_      0.04f                       /* 0.2^2 */

/* ---------------- 1. Farthest point sampling: one block per batch --------- */
__global__ void fps_kernel(const float* __restrict__ xyz, int* __restrict__ fidx) {
    const int b   = blockIdx.x;
    const int tid = threadIdx.x;                 // 256 threads, 16 pts each
    const float* xb = xyz + (size_t)b * 3 * N_;

    float px[16], py[16], pz[16], dist[16];
#pragma unroll
    for (int i = 0; i < 16; ++i) {
        int n = tid * 16 + i;
        px[i] = xb[n]; py[i] = xb[N_ + n]; pz[i] = xb[2 * N_ + n];
        dist[i] = 1e10f;
    }
    __shared__ float rv[256];
    __shared__ int   ri[256];
    __shared__ float scx, scy, scz;
    __shared__ int   sfar;
    if (tid == 0) sfar = 0;
    __syncthreads();

    for (int it = 0; it < NPOINT_; ++it) {
        if (tid == 0) {
            int f = sfar;
            fidx[b * NPOINT_ + it] = f;          // emit current 'far' (scan semantics)
            scx = xb[f]; scy = xb[N_ + f]; scz = xb[2 * N_ + f];
        }
        __syncthreads();
        float cx = scx, cy = scy, cz = scz;
        float bestv = -1.0f; int besti = 0;
#pragma unroll
        for (int i = 0; i < 16; ++i) {
            float dx = px[i] - cx, dy = py[i] - cy, dz = pz[i] - cz;
            float d  = dx * dx + dy * dy + dz * dz;
            float nd = fminf(dist[i], d);
            dist[i]  = nd;
            if (nd > bestv) { bestv = nd; besti = tid * 16 + i; }  // first-max in chunk
        }
        rv[tid] = bestv; ri[tid] = besti;
        __syncthreads();
        for (int off = 128; off > 0; off >>= 1) {
            if (tid < off) {
                float v2 = rv[tid + off]; int i2 = ri[tid + off];
                if (v2 > rv[tid] || (v2 == rv[tid] && i2 < ri[tid])) {
                    rv[tid] = v2; ri[tid] = i2;  // argmax, ties -> lower index
                }
            }
            __syncthreads();
        }
        if (tid == 0) sfar = ri[0];
        __syncthreads();
    }
}

/* ---------------- 2. Gather sampled centroids + emit new_xyz output ------- */
__global__ void gather_newxyz_kernel(const float* __restrict__ xyz, const int* __restrict__ fidx,
                                     float* __restrict__ nxyz, float* __restrict__ out) {
    int t = blockIdx.x * blockDim.x + threadIdx.x;   // b*NPOINT + s
    if (t >= B_ * NPOINT_) return;
    int b = t >> 10, s = t & 1023;
    int j = fidx[t];
#pragma unroll
    for (int c = 0; c < 3; ++c) {
        float v = xyz[(size_t)b * 3 * N_ + c * N_ + j];
        nxyz[t * 3 + c] = v;
        out[(size_t)b * 3 * NPOINT_ + c * NPOINT_ + s] = v;   // (B,3,NPOINT)
    }
}

/* ---------------- 3. Ball query: first NSAMPLE hits in ascending index ---- */
__global__ void query_ball_kernel(const float* __restrict__ xyz, const float* __restrict__ nxyz,
                                  int* __restrict__ idxb) {
    int t = blockIdx.x * blockDim.x + threadIdx.x;   // centroid id
    if (t >= B_ * NPOINT_) return;
    int b = t >> 10;
    const float* xb = xyz + (size_t)b * 3 * N_;
    float cx = nxyz[t * 3], cy = nxyz[t * 3 + 1], cz = nxyz[t * 3 + 2];
    int cnt = 0, f0 = 0;
    for (int n = 0; n < N_; ++n) {
        float dx = xb[n] - cx, dy = xb[N_ + n] - cy, dz = xb[2 * N_ + n] - cz;
        float d = dx * dx + dy * dy + dz * dz;
        if (d <= R2_) {
            if (cnt == 0) f0 = n;
            idxb[t * NSAMPLE_ + cnt] = n;
            if (++cnt == NSAMPLE_) break;
        }
    }
    for (int k = cnt; k < NSAMPLE_; ++k) idxb[t * NSAMPLE_ + k] = f0;   // pad with first
}

/* ---------------- 4. Weights -> bf16, zero-padded K ----------------------- */
__global__ void prep_weights_kernel(const float* __restrict__ w0, const float* __restrict__ w1,
                                    const float* __restrict__ w2, __bf16* __restrict__ wp) {
    int t = blockIdx.x * blockDim.x + threadIdx.x;
    if (t < 64 * 96) {                             // layer0: 64x67 -> 64x96
        int r = t / 96, c = t - r * 96;
        wp[t] = (c < 67) ? (__bf16)w0[r * 67 + c] : (__bf16)0.0f;
    } else if (t < 64 * 96 + 64 * 64) {            // layer1: 64x64
        wp[t] = (__bf16)w1[t - 64 * 96];
    } else if (t < 64 * 96 + 64 * 64 + 128 * 64) { // layer2: 128x64
        wp[t] = (__bf16)w2[t - (64 * 96 + 64 * 64)];
    }
}

/* ---------------- 5. Gather grouped features, point-major bf16 [n][K0] ---- */
__global__ void gather_feat_kernel(const float* __restrict__ xyz, const float* __restrict__ pts,
                                   const float* __restrict__ nxyz, const int* __restrict__ idxb,
                                   __bf16* __restrict__ x) {
    int t = blockIdx.x * blockDim.x + threadIdx.x;   // sample id in [0, NTOT)
    if (t >= NTOT) return;
    int bs = t >> 5;                                 // b*NPOINT + s
    int b  = bs >> 10;
    int j  = idxb[t];
    __bf16* xr = x + (size_t)t * K0_;
#pragma unroll
    for (int c = 0; c < 3; ++c)
        xr[c] = (__bf16)(xyz[(size_t)b * 3 * N_ + c * N_ + j] - nxyz[bs * 3 + c]);
    const float* pb = pts + (size_t)b * 64 * N_;
    for (int c = 0; c < 64; ++c)
        xr[3 + c] = (__bf16)pb[(size_t)c * N_ + j];
#pragma unroll
    for (int c = 67; c < K0_; ++c) xr[c] = (__bf16)0.0f;
}

/* ---------------- 6. WMMA GEMM, one column tile per wave, ALL M tiles ------ */
/* h[M][NTOT] = W[M][K] * X[NTOT][K]^T + bias. Each wave loads a B fragment    */
/* once per k-step and feeds MT WMMAs (register-resident accumulators), then   */
/* emits per-tile BatchNorm partial sums via a fixed shfl_xor butterfly.       */
template <int MT, int K>
__global__ void gemm_wmma_kernel(const __bf16* __restrict__ x, const __bf16* __restrict__ w,
                                 const float* __restrict__ bias, float* __restrict__ h,
                                 float* __restrict__ psum, float* __restrict__ psumsq) {
    int ntile = blockIdx.x * 8 + (threadIdx.x >> 5);
    int lane  = threadIdx.x & 31;
    int nlo = lane & 15;
    int hi  = lane >> 4;

    const __bf16* xrow = x + (size_t)(ntile * 16 + nlo) * K + hi * 16;  // B: col nlo, K base hi*16

    v8f acc[MT];
#pragma unroll
    for (int m = 0; m < MT; ++m) acc[m] = {};

#pragma unroll
    for (int k0 = 0; k0 < K; k0 += 32) {
        // B 32x16 bf16: lane col nlo holds K = hi*16 .. hi*16+15 contiguous (point-major)
        v16bf bm = *(const v16bf*)(xrow + k0);
#pragma unroll
        for (int m = 0; m < MT; ++m) {
            // A 16x32 bf16: elems 0..15 = K[k0+hi*16 .. k0+hi*16+15] contiguous
            const __bf16* wrow = w + (size_t)(m * 16 + nlo) * K + k0 + hi * 16;
            v8bf alo = *(const v8bf*)(wrow);
            v8bf ahi = *(const v8bf*)(wrow + 8);
            v16bf a  = __builtin_shufflevector(alo, ahi, 0, 1, 2, 3, 4, 5, 6, 7,
                                                         8, 9, 10, 11, 12, 13, 14, 15);
            acc[m] = __builtin_amdgcn_wmma_f32_16x16x32_bf16(false, a, false, bm,
                                                             (short)0, acc[m], false, false);
        }
    }

    int col = ntile * 16 + nlo;
#pragma unroll
    for (int m = 0; m < MT; ++m) {
#pragma unroll
        for (int r = 0; r < 8; ++r) {
            int ch = m * 16 + hi * 8 + r;            // output channel for this D element
            float v = acc[m][r] + bias[ch];
            h[(size_t)ch * NTOT + col] = v;
            // per-tile BN partials: reduce over the 16 lanes of this half-wave
            float s = v, s2 = v * v;
#pragma unroll
            for (int d = 1; d < 16; d <<= 1) {
                s  += __shfl_xor(s,  d, 32);
                s2 += __shfl_xor(s2, d, 32);
            }
            if (nlo == 0) {
                psum  [(size_t)ch * NTILES + ntile] = s;
                psumsq[(size_t)ch * NTILES + ntile] = s2;
            }
        }
    }
}

/* ---------------- 7. Reduce BN partials -> fused scale/shift --------------- */
__global__ void stats_reduce_kernel(const float* __restrict__ psum, const float* __restrict__ psumsq,
                                    float* __restrict__ scale, float* __restrict__ shift,
                                    const float* __restrict__ g, const float* __restrict__ bt) {
    int c = blockIdx.x, tid = threadIdx.x;
    const float* ps  = psum   + (size_t)c * NTILES;
    const float* ps2 = psumsq + (size_t)c * NTILES;
    float s = 0.f, s2 = 0.f;
    for (int i = tid; i < NTILES; i += 256) { s += ps[i]; s2 += ps2[i]; }
    __shared__ float sh[256], sh2[256];
    sh[tid] = s; sh2[tid] = s2;
    __syncthreads();
    for (int off = 128; off > 0; off >>= 1) {
        if (tid < off) { sh[tid] += sh[tid + off]; sh2[tid] += sh2[tid + off]; }
        __syncthreads();
    }
    if (tid == 0) {
        float m   = sh[0] * (1.0f / NTOT);
        float var = sh2[0] * (1.0f / NTOT) - m * m;
        float sc  = g[c] * rsqrtf(var + EPS_);
        scale[c] = sc;
        shift[c] = bt[c] - m * sc;
    }
}

/* ---------------- 8. BN + ReLU, repack point-major bf16 [n][64] ----------- */
__global__ void bn_relu_pack_kernel(const float* __restrict__ h, const float* __restrict__ scale,
                                    const float* __restrict__ shift, __bf16* __restrict__ xo) {
    int n = blockIdx.x * blockDim.x + threadIdx.x;
    if (n >= NTOT) return;
    __bf16* xr = xo + (size_t)n * 64;
#pragma unroll
    for (int c = 0; c < 64; ++c) {
        float v = h[(size_t)c * NTOT + n] * scale[c] + shift[c];
        xr[c] = (__bf16)fmaxf(v, 0.0f);
    }
}

/* ---------------- 9. BN + ReLU + max over nsample -> new_points ----------- */
__global__ void final_max_kernel(const float* __restrict__ h, const float* __restrict__ scale,
                                 const float* __restrict__ shift, float* __restrict__ out) {
    int t = blockIdx.x * blockDim.x + threadIdx.x;   // b*128*1024 + c*1024 + s
    if (t >= B_ * 128 * NPOINT_) return;
    int s = t & 1023;
    int c = (t >> 10) & 127;
    int b = t >> 17;
    const float* row = h + (size_t)c * NTOT + (size_t)(b * NPOINT_ + s) * NSAMPLE_;
    float sc = scale[c], sf = shift[c];
    float mx = 0.0f;                                  // relu outputs are >= 0
#pragma unroll
    for (int k = 0; k < NSAMPLE_; ++k)
        mx = fmaxf(mx, fmaxf(row[k] * sc + sf, 0.0f));
    out[t] = mx;
}

/* ---------------- launch --------------------------------------------------- */
extern "C" void kernel_launch(void* const* d_in, const int* in_sizes, int n_in,
                              void* d_out, int out_size, void* d_ws, size_t ws_size,
                              hipStream_t stream) {
    const float* xyz = (const float*)d_in[0];
    const float* pts = (const float*)d_in[1];
    const float* w0 = (const float*)d_in[2];  const float* b0 = (const float*)d_in[3];
    const float* g0 = (const float*)d_in[4];  const float* bt0 = (const float*)d_in[5];
    const float* w1 = (const float*)d_in[6];  const float* b1 = (const float*)d_in[7];
    const float* g1 = (const float*)d_in[8];  const float* bt1 = (const float*)d_in[9];
    const float* w2 = (const float*)d_in[10]; const float* b2 = (const float*)d_in[11];
    const float* g2 = (const float*)d_in[12]; const float* bt2 = (const float*)d_in[13];
    float* out = (float*)d_out;

    char* ws = (char*)d_ws;
    int*    fidx  = (int*)   (ws + 0);             // 64 KB
    float*  nxyz  = (float*) (ws + 65536);         // 192 KB
    int*    idxb  = (int*)   (ws + 262144);        // 2 MB
    __bf16* wp    = (__bf16*)(ws + 2359296);       // 36 KB (padded weights)
    float*  scale = (float*) (ws + 2396160);       // 512 B
    float*  shift = (float*) (ws + 2396672);       // 512 B
    __bf16* xbuf  = (__bf16*)(ws + 2397184);       // NTOT*96 bf16 = 96 MB (reused K=64)
    float*  h     = (float*) (ws + 103060480);     // 128*NTOT f32 = 256 MB
    float*  ps    = (float*) (ws + 371495936);     // 128*NTILES f32 = 16 MB
    float*  ps2   = (float*) (ws + 388273152);     // 128*NTILES f32 = 16 MB

    const __bf16* wp0 = wp;
    const __bf16* wp1 = wp + 64 * 96;
    const __bf16* wp2 = wp + 64 * 96 + 64 * 64;

    fps_kernel<<<B_, 256, 0, stream>>>(xyz, fidx);
    gather_newxyz_kernel<<<(B_ * NPOINT_ + 255) / 256, 256, 0, stream>>>(xyz, fidx, nxyz, out);
    query_ball_kernel<<<(B_ * NPOINT_ + 255) / 256, 256, 0, stream>>>(xyz, nxyz, idxb);
    prep_weights_kernel<<<(64 * 96 + 64 * 64 + 128 * 64 + 255) / 256, 256, 0, stream>>>(w0, w1, w2, wp);
    gather_feat_kernel<<<(NTOT + 255) / 256, 256, 0, stream>>>(xyz, pts, nxyz, idxb, xbuf);

    // layer 0: 96 -> 64
    gemm_wmma_kernel<4, 96><<<NTILES / 8, 256, 0, stream>>>(xbuf, wp0, b0, h, ps, ps2);
    stats_reduce_kernel<<<64, 256, 0, stream>>>(ps, ps2, scale, shift, g0, bt0);
    bn_relu_pack_kernel<<<(NTOT + 255) / 256, 256, 0, stream>>>(h, scale, shift, xbuf);

    // layer 1: 64 -> 64
    gemm_wmma_kernel<4, 64><<<NTILES / 8, 256, 0, stream>>>(xbuf, wp1, b1, h, ps, ps2);
    stats_reduce_kernel<<<64, 256, 0, stream>>>(ps, ps2, scale, shift, g1, bt1);
    bn_relu_pack_kernel<<<(NTOT + 255) / 256, 256, 0, stream>>>(h, scale, shift, xbuf);

    // layer 2: 64 -> 128
    gemm_wmma_kernel<8, 64><<<NTILES / 8, 256, 0, stream>>>(xbuf, wp2, b2, h, ps, ps2);
    stats_reduce_kernel<<<128, 256, 0, stream>>>(ps, ps2, scale, shift, g2, bt2);
    final_max_kernel<<<(B_ * 128 * NPOINT_ + 255) / 256, 256, 0, stream>>>(h, scale, shift,
                                                                           out + B_ * 3 * NPOINT_);
}
